// TransformerSelfAttnLayer_34643206209733
// MI455X (gfx1250) — matrix operand order repaired and checked by reading, not compiled, not run on hardware
//
#include <hip/hip_runtime.h>

#define WDIM 512
#define NB   32
#define CDIM 128
#define HDIM 8
#define DDIM 16

typedef __attribute__((ext_vector_type(16))) _Float16 v16h;
typedef __attribute__((ext_vector_type(8)))  _Float16 v8h;
typedef __attribute__((ext_vector_type(8)))  float    v8f;

static __device__ __forceinline__ v8f wmma32(v16h a, v16h b, v8f c) {
  return __builtin_amdgcn_wmma_f32_16x16x32_f16(false, a, false, b, (short)0, c, false, false);
}
static __device__ __forceinline__ v16h vcat(v8h lo, v8h hi) {
  v16h r;
#pragma unroll
  for (int i = 0; i < 8; ++i) { r[i] = lo[i]; r[i + 8] = hi[i]; }
  return r;
}
static __device__ __forceinline__ v8h vzero8() {
  v8h r;
#pragma unroll
  for (int i = 0; i < 8; ++i) r[i] = (_Float16)0.f;
  return r;
}
static __device__ __forceinline__ v16h vzero16() {
  v16h r;
#pragma unroll
  for (int i = 0; i < 16; ++i) r[i] = (_Float16)0.f;
  return r;
}
// A fragment (16x32 f16, row-major source, contiguous K): lane<16 -> K {kc..kc+7, kc+16..23};
// lane>=16 -> K {kc+8..15, kc+24..31}
static __device__ __forceinline__ v16h load_a(const _Float16* rowp, int kc, int half) {
  const _Float16* p = rowp + kc + half * 8;
  return vcat(*(const v8h*)p, *(const v8h*)(p + 16));
}
// B fragment (32x16 f16): B[k][n] stored as row n contiguous in k. lane<16 -> K {kc..kc+7, kc+16..23};
// lane>=16 -> K {kc+8..15, kc+24..31}
static __device__ __forceinline__ v16h load_b(const _Float16* colrow, int kc, int half) {
  const _Float16* p = colrow + kc + half * 8;
  return vcat(*(const v8h*)p, *(const v8h*)(p + 16));
}

// ---------------- LayerNorm -> f16 ----------------
__global__ void k_ln(const float* __restrict__ feat, const float* __restrict__ g,
                     const float* __restrict__ bb, _Float16* __restrict__ x16) {
  int row = blockIdx.x;
  int c = threadIdx.x;
  float v = feat[(size_t)row * CDIM + c];
  __shared__ float s1[CDIM], s2[CDIM];
  s1[c] = v; s2[c] = v * v;
  __syncthreads();
  for (int off = 64; off >= 1; off >>= 1) {
    if (c < off) { s1[c] += s1[c + off]; s2[c] += s2[c + off]; }
    __syncthreads();
  }
  float mu  = s1[0] * (1.f / CDIM);
  float var = s2[0] * (1.f / CDIM) - mu * mu;
  float y = (v - mu) * rsqrtf(var + 1e-5f) * g[c] + bb[c];
  x16[(size_t)row * CDIM + c] = (_Float16)y;
}

// ---------------- f32 -> f16 convert ----------------
__global__ void k_cvt(const float* __restrict__ src, _Float16* __restrict__ dst, int n) {
  int i = blockIdx.x * blockDim.x + threadIdx.x;
  if (i < n) dst[i] = (_Float16)src[i];
}

// ---------------- QKV projection: qkv = LN(x) @ Win^T + b ----------------
// Writes qkv16 [16384,384] (q scaled by 0.25) and transposed V copy vT16 [N,H,d,W]
__global__ void __launch_bounds__(256) k_qkv(const _Float16* __restrict__ x16,
                                             const _Float16* __restrict__ w16,
                                             const float* __restrict__ bias,
                                             _Float16* __restrict__ qkv16,
                                             _Float16* __restrict__ vT16) {
  const int NT = 24;  // 384/16 column tiles
  int wave = (blockIdx.x * blockDim.x + threadIdx.x) >> 5;
  int lane = threadIdx.x & 31;
  int r = lane & 15, half = lane >> 4;
  int mt = wave / NT, ct = wave % NT;
  int m0 = mt * 16, c0 = ct * 16;
  const _Float16* arow = x16 + (size_t)(m0 + r) * CDIM;
  const _Float16* brow = w16 + (size_t)(c0 + r) * CDIM;
  v8f acc = {};
#pragma unroll
  for (int kc = 0; kc < CDIM; kc += 32)
    acc = wmma32(load_a(arow, kc, half), load_b(brow, kc, half), acc);
  int cg = c0 + r;
  float bval = bias[cg];
  float scl = (cg < CDIM) ? 0.25f : 1.f;  // q *= d^-0.5
#pragma unroll
  for (int i = 0; i < 8; ++i) {
    int m = m0 + half * 8 + i;
    _Float16 hv = (_Float16)((acc[i] + bval) * scl);
    qkv16[(size_t)m * 384 + cg] = hv;
    if (cg >= 256) {  // V: also write [n,h,c,w] transposed copy for P@V B operand
      int cv = cg - 256, hh = cv >> 4, cd = cv & 15;
      int w = m >> 5, n = m & 31;
      vT16[(((size_t)n * HDIM + hh) * DDIM + cd) * WDIM + w] = hv;
    }
  }
}

// ---------------- PE projection: qr_kr = pos[idx] @ Win[:256]^T + b[:256] ----------------
// Output qr_kr [262144, 256] f16 (q_r = cols 0..127 scaled by 0.25, k_r = cols 128..255)
__global__ void __launch_bounds__(256) k_pe(const _Float16* __restrict__ pos16,
                                            const int* __restrict__ pidx,
                                            const _Float16* __restrict__ w16,
                                            const float* __restrict__ bias,
                                            _Float16* __restrict__ qr_kr) {
  const int NT = 16;  // 256/16 column tiles
  int wave = (blockIdx.x * blockDim.x + threadIdx.x) >> 5;
  int lane = threadIdx.x & 31;
  int r = lane & 15, half = lane >> 4;
  int mt = wave / NT, ct = wave % NT;
  int m0 = mt * 16, c0 = ct * 16;
  int idx = pidx[m0 + r];
  const _Float16* arow = pos16 + (size_t)idx * CDIM;
  const _Float16* brow = w16 + (size_t)(c0 + r) * CDIM;
  v8f acc = {};
#pragma unroll
  for (int kc = 0; kc < CDIM; kc += 32)
    acc = wmma32(load_a(arow, kc, half), load_b(brow, kc, half), acc);
  int cg = c0 + r;
  float bval = bias[cg];
  float scl = (cg < CDIM) ? 0.25f : 1.f;
#pragma unroll
  for (int i = 0; i < 8; ++i) {
    int m = m0 + half * 8 + i;
    qr_kr[(size_t)m * 256 + cg] = (_Float16)((acc[i] + bval) * scl);
  }
}

// ---------------- Flash attention with relative-position terms ----------------
// grid = (W/16/8, N, H); h is slowest -> all n for one head co-resident -> q_r/k_r head
// slice (16 MB) stays in the 192 MB L2 across the 32 batch replays.
__global__ void __launch_bounds__(256) k_attn(const _Float16* __restrict__ qkv16,
                                              const _Float16* __restrict__ qr_kr,
                                              const _Float16* __restrict__ vT16,
                                              _Float16* __restrict__ ao16) {
  __shared__ __align__(16) _Float16 lds[8][16 * 16];
  int wv = threadIdx.x >> 5;
  int lane = threadIdx.x & 31;
  int r = lane & 15, half = lane >> 4;
  int w0 = (blockIdx.x * 8 + wv) * 16;
  int n = blockIdx.y, h = blockIdx.z;

  // persistent q A-fragment (K=16 padded to 32 with zeros)
  const _Float16* qp = qkv16 + ((size_t)(w0 + r) * NB + n) * 384 + h * DDIM + half * 8;
  v8h qlo = *(const v8h*)qp;
  v8h z8 = vzero8();
  v16h Aq = vcat(qlo, z8);
  v16h Z16 = vzero16();

  float mi[8], li[8];
  v8f O = {};
#pragma unroll
  for (int i = 0; i < 8; ++i) { mi[i] = -1e30f; li[i] = 0.f; }

  for (int vt = 0; vt < WDIM / 16; ++vt) {
    int v0 = vt * 16;
    // per-lane k row (column r of the tile), contiguous 16 halves
    const _Float16* kp = qkv16 + ((size_t)(v0 + r) * NB + n) * 384 + CDIM + h * DDIM;
    v16h kfull = vcat(*(const v8h*)kp, *(const v8h*)(kp + 8));

    // term1: q @ k^T   (K=16, B high half zero)
    v8f S = {};
    S = wmma32(Aq, vcat(*(const v8h*)(kp + half * 8), z8), S);

    // term2: q . k_r  — K'=(w',c)=256: block-diagonal A built from own q frag, dense B from k_r
#pragma unroll
    for (int j = 0; j < 8; ++j) {
      int wA = w0 + 2 * j;
      const _Float16* krp = qr_kr + ((size_t)wA * WDIM + (v0 + r)) * 256 + CDIM + h * DDIM + half * 8;
      v16h B2 = vcat(*(const v8h*)krp, *(const v8h*)(krp + (size_t)WDIM * 256));
      v16h A2 = (r == 2 * j) ? vcat(qlo, z8) : ((r == 2 * j + 1) ? vcat(z8, qlo) : Z16);
      S = wmma32(A2, B2, S);
    }
    // term3: k . q_r  — K'=(v',c)=256: dense A from q_r, block-diagonal B from resident k row
    v8f S3 = {};
#pragma unroll
    for (int j = 0; j < 8; ++j) {
      const _Float16* qa =
          qr_kr + ((size_t)(w0 + r) * WDIM + (v0 + 2 * j + half)) * 256 + h * DDIM;
      v16h A3 = vcat(*(const v8h*)qa, *(const v8h*)(qa + 8));
      bool act = ((r >> 1) == j) && ((r & 1) == half);
      S3 = wmma32(A3, act ? kfull : Z16, S3);
    }
#pragma unroll
    for (int i = 0; i < 8; ++i) S[i] += S3[i];

    // online softmax; rows m = half*8+i live across the 16-lane half-wave
    float mn[8];
#pragma unroll
    for (int i = 0; i < 8; ++i) mn[i] = S[i];
#pragma unroll
    for (int off = 8; off >= 1; off >>= 1)
#pragma unroll
      for (int i = 0; i < 8; ++i) mn[i] = fmaxf(mn[i], __shfl_xor(mn[i], off));
    float p[8], rs[8];
#pragma unroll
    for (int i = 0; i < 8; ++i) {
      mn[i] = fmaxf(mn[i], mi[i]);
      float sc = __expf(mi[i] - mn[i]);
      p[i] = __expf(S[i] - mn[i]);
      li[i] *= sc;
      O[i] *= sc;
      mi[i] = mn[i];
      rs[i] = p[i];
    }
#pragma unroll
    for (int off = 8; off >= 1; off >>= 1)
#pragma unroll
      for (int i = 0; i < 8; ++i) rs[i] += __shfl_xor(rs[i], off);
#pragma unroll
    for (int i = 0; i < 8; ++i) li[i] += rs[i];

    // transpose P from C/D layout to A layout through LDS
    __syncthreads();
#pragma unroll
    for (int i = 0; i < 8; ++i) lds[wv][(half * 8 + i) * 16 + r] = (_Float16)p[i];
    __syncthreads();
    v8h plo = *(const v8h*)&lds[wv][r * 16 + half * 8];
    v16h Ap = vcat(plo, z8);
    // V B-operand from transposed copy: contiguous over v
    const _Float16* vtp = vT16 + (((size_t)n * HDIM + h) * DDIM + r) * WDIM + v0;
    v16h Bv = vcat(*(const v8h*)(vtp + half * 8), z8);
    O = wmma32(Ap, Bv, O);
  }
#pragma unroll
  for (int i = 0; i < 8; ++i) {
    int w = w0 + half * 8 + i;
    ao16[((size_t)w * NB + n) * CDIM + h * DDIM + r] = (_Float16)(O[i] / li[i]);
  }
}

// ---------------- Output projection + bias + residual ----------------
__global__ void __launch_bounds__(256) k_out(const _Float16* __restrict__ ao16,
                                             const _Float16* __restrict__ ow16,
                                             const float* __restrict__ ob,
                                             const float* __restrict__ feat,
                                             float* __restrict__ out) {
  const int NT = 8;  // 128/16
  int wave = (blockIdx.x * blockDim.x + threadIdx.x) >> 5;
  int lane = threadIdx.x & 31;
  int r = lane & 15, half = lane >> 4;
  int mt = wave / NT, ct = wave % NT;
  int m0 = mt * 16, c0 = ct * 16;
  const _Float16* arow = ao16 + (size_t)(m0 + r) * CDIM;
  const _Float16* brow = ow16 + (size_t)(c0 + r) * CDIM;
  v8f acc = {};
#pragma unroll
  for (int kc = 0; kc < CDIM; kc += 32)
    acc = wmma32(load_a(arow, kc, half), load_b(brow, kc, half), acc);
  int cg = c0 + r;
  float bval = ob[cg];
#pragma unroll
  for (int i = 0; i < 8; ++i) {
    int m = m0 + half * 8 + i;
    out[(size_t)m * CDIM + cg] = acc[i] + bval + feat[(size_t)m * CDIM + cg];
  }
}

extern "C" void kernel_launch(void* const* d_in, const int* in_sizes, int n_in,
                              void* d_out, int out_size, void* d_ws, size_t ws_size,
                              hipStream_t stream) {
  (void)in_sizes; (void)n_in; (void)out_size;
  const float* feat  = (const float*)d_in[0];
  const float* pos   = (const float*)d_in[1];
  const int*   pidx  = (const int*)d_in[2];
  const float* in_w  = (const float*)d_in[3];
  const float* in_b  = (const float*)d_in[4];
  const float* out_w = (const float*)d_in[5];
  const float* out_b = (const float*)d_in[6];
  const float* ln_g  = (const float*)d_in[7];
  const float* ln_b  = (const float*)d_in[8];

  char* ws = (char*)d_ws;
  size_t off = 0;
  auto alloc = [&](size_t bytes) -> void* {
    void* p = ws + off;
    off += (bytes + 255) & ~(size_t)255;
    return p;
  };
  _Float16* x16    = (_Float16*)alloc((size_t)16384 * 128 * 2);           // 4 MB
  _Float16* qkv16  = (_Float16*)alloc((size_t)16384 * 384 * 2);           // 12 MB
  _Float16* vT16   = (_Float16*)alloc((size_t)NB * HDIM * DDIM * WDIM * 2); // 4 MB
  _Float16* qrkr   = (_Float16*)alloc((size_t)262144 * 256 * 2);          // 134 MB
  _Float16* ao16   = (_Float16*)alloc((size_t)16384 * 128 * 2);           // 4 MB
  _Float16* pos16  = (_Float16*)alloc((size_t)1023 * 128 * 2);
  _Float16* w_in16 = (_Float16*)alloc((size_t)384 * 128 * 2);
  _Float16* ow16   = (_Float16*)alloc((size_t)128 * 128 * 2);
  if (off > ws_size) return;  // workspace too small: do nothing (deterministic)

  k_ln<<<16384, 128, 0, stream>>>(feat, ln_g, ln_b, x16);
  k_cvt<<<(1023 * 128 + 255) / 256, 256, 0, stream>>>(pos, pos16, 1023 * 128);
  k_cvt<<<(384 * 128 + 255) / 256, 256, 0, stream>>>(in_w, w_in16, 384 * 128);
  k_cvt<<<(128 * 128 + 255) / 256, 256, 0, stream>>>(out_w, ow16, 128 * 128);
  k_qkv<<<(1024 * 24) / 8, 256, 0, stream>>>(x16, w_in16, in_b, qkv16, vT16);
  k_pe<<<(16384 * 16) / 8, 256, 0, stream>>>(pos16, pidx, w_in16, in_b, qrkr);
  dim3 ag(WDIM / 16 / 8, NB, HDIM);  // h slowest for L2 reuse of qr/kr head slices
  k_attn<<<ag, 256, 0, stream>>>(qkv16, qrkr, vT16, ao16);
  k_out<<<(1024 * 8) / 8, 256, 0, stream>>>(ao16, ow16, out_b, feat, (float*)d_out);
}
